// BiLSTM_CRF_80693845557556
// MI455X (gfx1250) — compile-verified, matrix operand order checked by reading, not run on hardware
//
#include <hip/hip_runtime.h>
#include <math.h>

// ---------------- problem constants ----------------
constexpr int SEQ_T   = 512;
constexpr int BATCH_N = 64;
constexpr int EMB     = 256;
constexpr int HID     = 256;
constexpr int NSTATE  = 64;
constexpr int TBROWS  = SEQ_T * BATCH_N;   // 32768
constexpr int G4      = 4 * HID;           // 1024 gate columns
constexpr int FEAT2   = 2 * HID;           // 512

// ---------------- CDNA5 WMMA types ----------------
typedef __bf16 bf16;
typedef __attribute__((ext_vector_type(8)))  __bf16 v8bf;
typedef __attribute__((ext_vector_type(16))) __bf16 v16bf;
typedef __attribute__((ext_vector_type(8)))  float  v8f;

// Load a 16x32 bf16 fragment (A layout; also B when weights stored [N,K] row-major).
// ISA 16-bit A layout: lane L holds row (L&15), K chunks [(L>>4)*8 .. +7] and [+16 .. +23].
__device__ __forceinline__ v16bf load_frag_nk(const bf16* p, int ld, int lane) {
  const int r  = lane & 15;
  const int kc = (lane >> 4) << 3;
  const bf16* q = p + (size_t)r * ld + kc;
  v8bf lo = *reinterpret_cast<const v8bf*>(q);
  v8bf hi = *reinterpret_cast<const v8bf*>(q + 16);
  return __builtin_shufflevector(lo, hi, 0,1,2,3,4,5,6,7,8,9,10,11,12,13,14,15);
}

__device__ __forceinline__ v8f wmma_bf16(v16bf a, v16bf b, v8f c) {
  // D = A(16x32) * B(32x16) + C, f32 accumulate
  return __builtin_amdgcn_wmma_f32_16x16x32_bf16(false, a, false, b, (short)0, c,
                                                 false, false);
}

__device__ __forceinline__ float sigmoidf(float x) {
  return 1.0f / (1.0f + __expf(-x));
}

// ---------------- kernel 1: f32 -> bf16 weight conversion ----------------
__global__ void cvt_bf16_kernel(const float* __restrict__ src, bf16* __restrict__ dst, int n) {
  int i = blockIdx.x * blockDim.x + threadIdx.x;
  if (i < n) dst[i] = (bf16)src[i];
}

// ---------------- kernel 2: embedding gather + bf16 convert ----------------
__global__ void gather_embed_kernel(const int* __restrict__ obs,
                                    const float* __restrict__ embed,
                                    bf16* __restrict__ xb) {
  const int row = blockIdx.x;          // t*BATCH_N + b
  const int e   = threadIdx.x;         // 0..255
  const int idx = obs[row];
  xb[(size_t)row * EMB + e] = (bf16)embed[(size_t)idx * EMB + e];
}

// ---------------- kernel 3: input projection GEMM ----------------
// gatesT[dir][t][col][batch] = bias[col] + (xb[t*64+b,:] . w_ih[dir][col,:])
// Transposed (col-major-in-batch) output layout: each lane's 8 D values
// (fixed N=col, M=batch contiguous) become ONE b128 store, and the sequential
// recurrent kernel re-loads them with ONE b128 load per gate tile.
__global__ void input_gemm_kernel(const bf16* __restrict__ xb,
                                  const bf16* __restrict__ wihb,
                                  const float* __restrict__ b_f,
                                  const float* __restrict__ b_b,
                                  bf16* __restrict__ gatesT) {
  const int lane = threadIdx.x & 31;
  const int wave = threadIdx.x >> 5;
  int tile = blockIdx.x * 8 + wave;          // 0 .. 262143
  const int dir = tile >> 17;                // 131072 tiles per direction
  tile &= 131071;
  const int mt = tile >> 6;                  // 0..2047 (16 TB-rows each; within one t)
  const int nt = tile & 63;                  // 0..63
  const bf16* A0 = xb + (size_t)mt * 16 * EMB;
  const bf16* B0 = wihb + (size_t)dir * G4 * EMB + (size_t)nt * 16 * EMB;
  const int n = lane & 15, mh = (lane >> 4) << 3;
  const int col = nt * 16 + n;
  const float bv = (dir ? b_b : b_f)[col];   // fold bias in here (f32 accum)
  v8f acc = {bv, bv, bv, bv, bv, bv, bv, bv};
#pragma unroll
  for (int k0 = 0; k0 < EMB; k0 += 32) {
    v16bf a = load_frag_nk(A0 + k0, EMB, lane);
    v16bf b = load_frag_nk(B0 + k0, EMB, lane);
    acc = wmma_bf16(a, b, acc);
  }
  // transposed store: t = mt/4, batch base b0 = (mt%4)*16 + mh, 8 contiguous
  const int t  = mt >> 2;
  const int b0 = ((mt & 3) << 4) + mh;
  bf16* gout = gatesT + ((size_t)dir * SEQ_T + t) * G4 * BATCH_N;
  v8bf ob;
#pragma unroll
  for (int r = 0; r < 8; ++r) ob[r] = (bf16)acc[r];
  *reinterpret_cast<v8bf*>(&gout[(size_t)col * BATCH_N + b0]) = ob;
}

// ---------------- kernel 4: sequential LSTM recurrence ----------------
// grid = 2 (one block per direction), block = 1024 (32 waves)
// h in LDS (bf16), c entirely in registers, W_hh from global (L2 resident).
// Input-gate preactivations come in transposed layout -> one b128 load per gate.
__global__ void lstm_recurrent_kernel(const bf16* __restrict__ gatesT,
                                      const bf16* __restrict__ whhb,
                                      const float* __restrict__ h0,
                                      const float* __restrict__ c0,
                                      bf16* __restrict__ feats) {
  const int dir = blockIdx.x;
  const bf16* gin = gatesT + (size_t)dir * SEQ_T * G4 * BATCH_N;
  const bf16* whh = whhb + (size_t)dir * G4 * HID;

  __shared__ __align__(16) bf16 h_lds[BATCH_N][HID];   // 32 KB
  for (int i = threadIdx.x; i < BATCH_N * HID; i += blockDim.x)
    h_lds[i >> 8][i & 255] = (bf16)h0[(size_t)dir * BATCH_N * HID + i];

  const int lane = threadIdx.x & 31;
  const int wave = threadIdx.x >> 5;
  const int n = lane & 15, mh = (lane >> 4) << 3;

  // each wave owns two (batch-tile, h-col-tile) pairs; cell state in registers
  int mts[2], hcs[2];
  float creg[2][8];
#pragma unroll
  for (int q = 0; q < 2; ++q) {
    int p = wave * 2 + q;        // 0..63
    mts[q] = p & 3;              // batch tile 0..3
    hcs[q] = p >> 2;             // h-column tile 0..15
#pragma unroll
    for (int r = 0; r < 8; ++r) {
      int m = mts[q] * 16 + mh + r;
      int col = hcs[q] * 16 + n;
      creg[q][r] = c0[(size_t)dir * BATCH_N * HID + (size_t)m * HID + col];
    }
  }
  __syncthreads();

  for (int step = 0; step < SEQ_T; ++step) {
    const int t = dir ? (SEQ_T - 1 - step) : step;
    const bf16* gt = gin + (size_t)t * G4 * BATCH_N;
    float hnew[2][8];
#pragma unroll
    for (int q = 0; q < 2; ++q) {
      const int mt = mts[q], hc = hcs[q];
      const int b0 = mt * 16 + mh;           // batch base, 8 contiguous values
      v8f acc[4];
      // init accumulators from transposed preactivations: ONE b128 load / gate
#pragma unroll
      for (int g = 0; g < 4; ++g) {
        const int col = g * HID + hc * 16 + n;
        const v8bf gv = *reinterpret_cast<const v8bf*>(&gt[(size_t)col * BATCH_N + b0]);
#pragma unroll
        for (int r = 0; r < 8; ++r) acc[g][r] = (float)gv[r];
      }
      // h @ W_hh^T : K = 256, 8 wmma steps x 4 gates
#pragma unroll
      for (int k0 = 0; k0 < HID; k0 += 32) {
        v16bf a = load_frag_nk(&h_lds[mt * 16][0] + k0, HID, lane);
#pragma unroll
        for (int g = 0; g < 4; ++g) {
          v16bf bm = load_frag_nk(whh + (size_t)(g * HID + hc * 16) * HID + k0, HID, lane);
          acc[g] = wmma_bf16(a, bm, acc[g]);
        }
      }
      // gate nonlinearity + cell update (torch gate order i,f,g,o)
#pragma unroll
      for (int r = 0; r < 8; ++r) {
        const float iv = sigmoidf(acc[0][r]);
        const float fv = sigmoidf(acc[1][r]);
        const float gv = tanhf(acc[2][r]);
        const float ov = sigmoidf(acc[3][r]);
        const float cn = fv * creg[q][r] + iv * gv;
        creg[q][r] = cn;
        hnew[q][r] = ov * tanhf(cn);
      }
    }
    __syncthreads();   // everyone done reading h_lds of this step
#pragma unroll
    for (int q = 0; q < 2; ++q) {
      const int mt = mts[q], hc = hcs[q];
#pragma unroll
      for (int r = 0; r < 8; ++r) {
        const int m = mt * 16 + mh + r;
        const int col = hc * 16 + n;
        const bf16 hb = (bf16)hnew[q][r];
        h_lds[m][col] = hb;
        feats[((size_t)t * BATCH_N + m) * FEAT2 + dir * HID + col] = hb;
      }
    }
    __syncthreads();   // h updated before next step reads it
  }
}

// ---------------- kernel 5: emission GEMM ----------------
// emis[TB][64] = feats[TB,512] @ lin_w.T + lin_b
__global__ void emis_gemm_kernel(const bf16* __restrict__ feats,
                                 const bf16* __restrict__ linwb,
                                 const float* __restrict__ lin_b,
                                 float* __restrict__ emis) {
  const int lane = threadIdx.x & 31;
  const int wave = threadIdx.x >> 5;
  const int tile = blockIdx.x * 8 + wave;    // 0..8191
  const int mt = tile >> 2;                  // 0..2047
  const int nt = tile & 3;                   // 0..3
  const bf16* A0 = feats + (size_t)mt * 16 * FEAT2;
  const bf16* B0 = linwb + (size_t)nt * 16 * FEAT2;
  const int n = lane & 15, mh = (lane >> 4) << 3;
  const float bv = lin_b[nt * 16 + n];
  v8f acc = {bv, bv, bv, bv, bv, bv, bv, bv};
#pragma unroll
  for (int k0 = 0; k0 < FEAT2; k0 += 32) {
    v16bf a = load_frag_nk(A0 + k0, FEAT2, lane);
    v16bf b = load_frag_nk(B0 + k0, FEAT2, lane);
    acc = wmma_bf16(a, b, acc);
  }
#pragma unroll
  for (int r = 0; r < 8; ++r) {
    size_t row = (size_t)mt * 16 + mh + r;
    emis[row * NSTATE + nt * 16 + n] = acc[r];
  }
}

// ---------------- kernel 6: Viterbi forward + backtrack ----------------
// single block, pairwise + double-buffered alpha in LDS, byte backpointers
__global__ void viterbi_kernel(const float* __restrict__ emis,
                               const float* __restrict__ pairwise,
                               const float* __restrict__ start,
                               const float* __restrict__ stopv,
                               unsigned char* __restrict__ bps,
                               float* __restrict__ out) {
  __shared__ float pw[NSTATE][NSTATE];       // 16 KB
  __shared__ float alpha[2][BATCH_N][NSTATE];// 32 KB
  const int tid = threadIdx.x;
  for (int i = tid; i < NSTATE * NSTATE; i += blockDim.x)
    pw[i >> 6][i & 63] = pairwise[i];
  for (int i = tid; i < BATCH_N * NSTATE; i += blockDim.x) {
    int s = i & 63;
    alpha[0][i >> 6][s] = emis[i] + start[s];
  }
  __syncthreads();

  int cur = 0;
  for (int t = 1; t < SEQ_T; ++t) {
    for (int i = tid; i < BATCH_N * NSTATE; i += blockDim.x) {
      const int b = i >> 6, s = i & 63;
      float best = -3.4e38f;
      int bi = 0;
#pragma unroll 8
      for (int sp = 0; sp < NSTATE; ++sp) {
        float v = alpha[cur][b][sp] + pw[sp][s];
        if (v > best) { best = v; bi = sp; }   // first-max, matches jnp.argmax
      }
      alpha[1 - cur][b][s] = best + emis[(size_t)t * BATCH_N * NSTATE + i];
      bps[(size_t)(t - 1) * BATCH_N * NSTATE + i] = (unsigned char)bi;
    }
    __syncthreads();
    cur ^= 1;
  }

  if (tid < BATCH_N) {
    const int b = tid;
    float best = -3.4e38f;
    int bi = 0;
    for (int s = 0; s < NSTATE; ++s) {
      float v = alpha[cur][b][s] + stopv[s];
      if (v > best) { best = v; bi = s; }
    }
    out[b] = best;                       // max_score
    float* seq = out + BATCH_N;          // state_seq [T,B] as float
    int st = bi;
    seq[(SEQ_T - 1) * BATCH_N + b] = (float)st;
    for (int t = SEQ_T - 2; t >= 0; --t) {
      st = bps[(size_t)t * BATCH_N * NSTATE + (size_t)b * NSTATE + st];
      seq[t * BATCH_N + b] = (float)st;
    }
  }
}

// ---------------- host launch ----------------
extern "C" void kernel_launch(void* const* d_in, const int* in_sizes, int n_in,
                              void* d_out, int out_size, void* d_ws, size_t ws_size,
                              hipStream_t stream) {
  const int*   obs      = (const int*)  d_in[0];
  const float* embed    = (const float*)d_in[1];
  const float* w_ih_f   = (const float*)d_in[2];
  const float* w_hh_f   = (const float*)d_in[3];
  const float* b_f      = (const float*)d_in[4];
  const float* w_ih_b   = (const float*)d_in[5];
  const float* w_hh_b   = (const float*)d_in[6];
  const float* b_b      = (const float*)d_in[7];
  const float* lin_w    = (const float*)d_in[8];
  const float* lin_b    = (const float*)d_in[9];
  const float* pairwise = (const float*)d_in[10];
  const float* start    = (const float*)d_in[11];
  const float* stopv    = (const float*)d_in[12];
  const float* h0       = (const float*)d_in[13];
  const float* c0       = (const float*)d_in[14];

  char* ws = (char*)d_ws;
  size_t off = 0;
  auto take = [&](size_t bytes) -> void* {
    void* p = ws + off;
    off += (bytes + 255) & ~(size_t)255;
    return p;
  };
  bf16* xb      = (bf16*)take((size_t)TBROWS * EMB * 2);        // 16 MB
  bf16* wihb    = (bf16*)take((size_t)2 * G4 * EMB * 2);        // 1 MB
  bf16* whhb    = (bf16*)take((size_t)2 * G4 * HID * 2);        // 1 MB
  bf16* linwb   = (bf16*)take((size_t)NSTATE * FEAT2 * 2);      // 64 KB
  bf16* gatesT  = (bf16*)take((size_t)2 * SEQ_T * G4 * BATCH_N * 2); // 128 MB
  bf16* feats   = (bf16*)take((size_t)TBROWS * FEAT2 * 2);      // 32 MB
  float* emis   = (float*)take((size_t)TBROWS * NSTATE * 4);    // 8 MB
  unsigned char* bps = (unsigned char*)take((size_t)(SEQ_T - 1) * BATCH_N * NSTATE); // 2 MB

  // weight conversion f32 -> bf16
  cvt_bf16_kernel<<<1024, 256, 0, stream>>>(w_ih_f, wihb,            G4 * EMB);
  cvt_bf16_kernel<<<1024, 256, 0, stream>>>(w_ih_b, wihb + G4 * EMB, G4 * EMB);
  cvt_bf16_kernel<<<1024, 256, 0, stream>>>(w_hh_f, whhb,            G4 * HID);
  cvt_bf16_kernel<<<1024, 256, 0, stream>>>(w_hh_b, whhb + G4 * HID, G4 * HID);
  cvt_bf16_kernel<<<128,  256, 0, stream>>>(lin_w,  linwb,           NSTATE * FEAT2);

  // embedding gather
  gather_embed_kernel<<<TBROWS, 256, 0, stream>>>(obs, embed, xb);

  // parallel input projection (both directions): 262144 tiles, 8 waves/block
  input_gemm_kernel<<<32768, 256, 0, stream>>>(xb, wihb, b_f, b_b, gatesT);

  // sequential recurrence: one workgroup per direction, concurrently
  lstm_recurrent_kernel<<<2, 1024, 0, stream>>>(gatesT, whhb, h0, c0, feats);

  // emission GEMM: 8192 tiles
  emis_gemm_kernel<<<1024, 256, 0, stream>>>(feats, linwb, lin_b, emis);

  // Viterbi + backtrack, writes [max_score | state_seq] to d_out
  viterbi_kernel<<<1, 1024, 0, stream>>>(emis, pairwise, start, stopv, bps, (float*)d_out);
}
